// PhysicsAwareIMUImputer_1975684956790
// MI455X (gfx1250) — matrix operand order repaired and checked by reading, not compiled, not run on hardware
//
#include <hip/hip_runtime.h>

// ---------------------------------------------------------------------------
// PhysicsAwareIMUImputer on MI455X (gfx1250, wave32, WMMA)
//
// Kernel 1 (recurrent): 8 blocks x 128 threads (4 waves). Each block owns a
// 16-row batch tile; all weights live in LDS as f16, every per-step GEMM is
// v_wmma_f32_16x16x32_f16 with f32 accumulation. Sequential scan over T=4096.
// Wave index is readfirstlane'd so N-tile loops are scalar (EXEC all-ones
// across every WMMA, per ISA 7.12 requirement) and cheap.
//
// Kernel 2 (heads): one thread per (b,t) row, head weights cached in LDS.
// ---------------------------------------------------------------------------

typedef __attribute__((ext_vector_type(16))) _Float16 v16h;
typedef __attribute__((ext_vector_type(8)))  float    v8f;

#define BATCH   128
#define TSTEPS  4096
#define INDIM   13

struct RecParams {
    const float* x;
    const float* wi; const float* wh; const float* lb;
    const float* c0w[4]; const float* c0b[4];
    const float* c1w[4]; const float* c1b[4];
    const float* c2w[4]; const float* c2b[4];
    float* cfc;                 // (B, T, 6) fp32 scratch
};

struct HeadParams {
    const float* gw1; const float* gb1; const float* gw2; const float* gb2;
    const float* aw1; const float* ab1; const float* aw2; const float* ab2;
    const float* uw1; const float* ub1; const float* uw2; const float* ub2;
    const float* cfc;
    float* out;                 // pred (B*T*6) then unc (B*T*6)
};

__device__ __forceinline__ float sigm(float x) { return 1.0f / (1.0f + __expf(-x)); }

// single-v_exp tanh; clamp keeps e2x finite so the ratio is well defined
__device__ __forceinline__ float ftanh(float x) {
    x = fminf(fmaxf(x, -10.0f), 10.0f);
    float e = __expf(2.0f * x);
    return (e - 1.0f) / (e + 1.0f);
}

__device__ __forceinline__ v8f wmma32(v16h a, v16h b, v8f c) {
    // D = A(16x32 f16) x B(32x16 f16) + C(16x16 f32)
    return __builtin_amdgcn_wmma_f32_16x16x32_f16(
        /*neg_a=*/false, a, /*neg_b=*/false, b,
        /*c_mod=*/(short)0, c, /*reuse_a=*/false, /*reuse_b=*/false);
}

// A-matrix (16x32, f16) lane layout: lane half hh, element i -> K = 16*(i/8) + 8*hh + (i%8)
__device__ __forceinline__ v16h ldA(const _Float16* Z, int ld, int m16, int hh, int kt) {
    const _Float16* p = Z + m16 * ld + kt * 32 + 8 * hh;
    v16h a;
#pragma unroll
    for (int i = 0; i < 8; ++i) { a[i] = p[i]; a[8 + i] = p[16 + i]; }
    return a;
}

// B-matrix (32x16, f16): lane n = lane%16, half hh, element i -> K = kt*32 + 16*(i/8) + 8*hh + (i%8)
// Weight stored row-major W[n][k] (B[k][n] == W[n][k] since we compute z @ W^T).
__device__ __forceinline__ v16h ldB(const _Float16* W, int ld, int m16, int hh, int kt, int nt) {
    const _Float16* q = W + (nt * 16 + m16) * ld + kt * 32 + 8 * hh;
    v16h b;
#pragma unroll
    for (int i = 0; i < 8; ++i) { b[i] = q[i]; b[8 + i] = q[16 + i]; }
    return b;
}

// C/D (16x16, f32): element r -> M = r + 8*hh, N = lane%16. Epilogue adds bias.
__device__ __forceinline__ void stC(float* G, const float* bias, int m16, int hh, int nt, v8f acc) {
    int nc = nt * 16 + m16;
    float bv = bias[nc];
#pragma unroll
    for (int r = 0; r < 8; ++r) G[(r + 8 * hh) * 256 + nc] = acc[r] + bv;
}

__global__ __launch_bounds__(128) void rec_kernel(RecParams P) {
    // f16 weights, zero-padded to K-tile(32)/N-tile(16) multiples
    __shared__ _Float16 sWl[256 * 96];   // LSTM [x|h] fused: N=256, K=77->96
    __shared__ _Float16 sW0[144 * 64];   // CfC0 [w1;w2;wa;wb]: N=140->144, K=48->64
    __shared__ _Float16 sW1[96 * 64];    // CfC1: N=92->96, K=58->64
    __shared__ _Float16 sW2[32 * 32];    // CfC2: N=24->32, K=29->32
    __shared__ float sBl[256], sB0[144], sB1[96], sB2[32];
    // activations / state (f16 inputs to WMMA, fp32 state & accumulators)
    __shared__ _Float16 zl[16 * 96];     // [x_t | h_prev | pad]
    __shared__ _Float16 z0[16 * 64];     // [x_t | h0_lstm | pad]
    __shared__ _Float16 z1[16 * 64];     // [out0 | h1_lstm | pad]
    __shared__ _Float16 z2[16 * 32];     // [out1 | h2_lstm | pad]
    __shared__ float    cst[16 * 64];    // LSTM cell state
    __shared__ float    gbuf[16 * 256];  // fp32 GEMM output staging

    const int tid  = threadIdx.x;
    const int lane = tid & 31;
    // wave id as SGPR -> scalar N-tile loops, EXEC stays all-ones across WMMA
    const int wid  = __builtin_amdgcn_readfirstlane(tid >> 5);
    const int m16  = lane & 15;
    const int hh   = lane >> 4;

    // ---- one-time weight conversion fp32 -> f16 into LDS ----
    for (int i = tid; i < 256 * 96; i += 128) {
        int n = i / 96, k = i % 96;
        float v = (k < 13) ? P.wi[n * 13 + k] : ((k < 77) ? P.wh[n * 64 + (k - 13)] : 0.0f);
        sWl[i] = (_Float16)v;
    }
    for (int i = tid; i < 144 * 64; i += 128) {
        int r = i / 64, k = i % 64; float v = 0.0f;
        if (r < 140 && k < 48) { int g = r / 35, j = r % 35; v = P.c0w[g][j * 48 + k]; }
        sW0[i] = (_Float16)v;
    }
    for (int i = tid; i < 96 * 64; i += 128) {
        int r = i / 64, k = i % 64; float v = 0.0f;
        if (r < 92 && k < 58) { int g = r / 23, j = r % 23; v = P.c1w[g][j * 58 + k]; }
        sW1[i] = (_Float16)v;
    }
    for (int i = tid; i < 32 * 32; i += 128) {
        int r = i / 32, k = i % 32; float v = 0.0f;
        if (r < 24 && k < 29) { int g = r / 6, j = r % 6; v = P.c2w[g][j * 29 + k]; }
        sW2[i] = (_Float16)v;
    }
    for (int i = tid; i < 256; i += 128) sBl[i] = P.lb[i];
    for (int i = tid; i < 144; i += 128) sB0[i] = (i < 140) ? P.c0b[i / 35][i % 35] : 0.0f;
    for (int i = tid; i < 96;  i += 128) sB1[i] = (i < 92)  ? P.c1b[i / 23][i % 23] : 0.0f;
    for (int i = tid; i < 32;  i += 128) sB2[i] = (i < 24)  ? P.c2b[i / 6][i % 6]   : 0.0f;
    // ---- state init (h0 = c0 = 0, pads = 0) ----
    for (int i = tid; i < 16 * 96; i += 128) zl[i] = (_Float16)0.0f;
    for (int i = tid; i < 16 * 64; i += 128) { z0[i] = (_Float16)0.0f; z1[i] = (_Float16)0.0f; cst[i] = 0.0f; }
    for (int i = tid; i < 16 * 32; i += 128) z2[i] = (_Float16)0.0f;
    __syncthreads();

    const float* xb   = P.x   + (size_t)(blockIdx.x * 16) * TSTEPS * INDIM;
    float*       cfcb = P.cfc + (size_t)(blockIdx.x * 16) * TSTEPS * 6;

    for (int t = 0; t < TSTEPS; ++t) {
        // load x_t into zl[:,0:13] and z0[:,0:13]
        for (int i = tid; i < 16 * INDIM; i += 128) {
            int m = i / INDIM, k = i % INDIM;
            float v = xb[(size_t)m * (TSTEPS * INDIM) + (size_t)t * INDIM + k];
            _Float16 hv = (_Float16)v;
            zl[m * 96 + k] = hv;
            z0[m * 64 + k] = hv;
        }
        if (tid < 16 && t + 1 < TSTEPS)   // global_prefetch_b8: next step's x rows
            __builtin_prefetch(xb + (size_t)tid * (TSTEPS * INDIM) + (size_t)(t + 1) * INDIM, 0, 1);
        __syncthreads();

        // ---- LSTM GEMM: zg(16x256) = [x|h](16x96) @ Wc^T, 3 K-tiles x 16 N-tiles ----
        {
            v16h A0 = ldA(zl, 96, m16, hh, 0);
            v16h A1 = ldA(zl, 96, m16, hh, 1);
            v16h A2 = ldA(zl, 96, m16, hh, 2);
#pragma unroll
            for (int q = 0; q < 4; ++q) {
                int nt = wid * 4 + q;                 // scalar (wid is SGPR)
                v8f acc = {0, 0, 0, 0, 0, 0, 0, 0};
                acc = wmma32(A0, ldB(sWl, 96, m16, hh, 0, nt), acc);
                acc = wmma32(A1, ldB(sWl, 96, m16, hh, 1, nt), acc);
                acc = wmma32(A2, ldB(sWl, 96, m16, hh, 2, nt), acc);
                stC(gbuf, sBl, m16, hh, nt, acc);
            }
        }
        __syncthreads();
        // ---- LSTM gates (ia|ig|fg|og), scatter h_lstm slices into z0/z1/z2 ----
        for (int i = tid; i < 16 * 64; i += 128) {
            int m = i >> 6, j = i & 63;
            const float* g = &gbuf[m * 256];
            float cn = cst[i] * sigm(g[128 + j] + 1.0f) + ftanh(g[j]) * sigm(g[64 + j]);
            float hv = ftanh(cn) * sigm(g[192 + j]);
            cst[i] = cn;
            _Float16 hf = (_Float16)hv;
            if (j < 35)      z0[m * 64 + 13 + j] = hf;          // h0 slice
            else if (j < 58) z1[m * 64 + j]      = hf;          // h1 slice at cols 35..57
            else             z2[m * 32 + 23 + (j - 58)] = hf;   // h2 slice at cols 23..28
        }
        __syncthreads();

        // ---- CfC0 GEMM: (16x48)->(16x140), 2 K-tiles x 9 N-tiles ----
        {
            v16h A0 = ldA(z0, 64, m16, hh, 0);
            v16h A1 = ldA(z0, 64, m16, hh, 1);
            for (int nt = wid; nt < 9; nt += 4) {   // scalar loop (wid is SGPR)
                v8f acc = {0, 0, 0, 0, 0, 0, 0, 0};
                acc = wmma32(A0, ldB(sW0, 64, m16, hh, 0, nt), acc);
                acc = wmma32(A1, ldB(sW0, 64, m16, hh, 1, nt), acc);
                stC(gbuf, sB0, m16, hh, nt, acc);
            }
        }
        __syncthreads();
        for (int i = tid; i < 16 * 35; i += 128) {
            int m = i / 35, j = i % 35;
            const float* g = &gbuf[m * 256];
            float f1 = ftanh(g[j]), f2 = ftanh(g[35 + j]);
            float tt = sigm(g[70 + j] + g[105 + j]);
            float o  = f1 * (1.0f - tt) + tt * f2;
            _Float16 of = (_Float16)o;
            z1[m * 64 + j]      = of;   // input to CfC1
            zl[m * 96 + 13 + j] = of;   // next-step LSTM h (cols 0..34)
        }
        __syncthreads();

        // ---- CfC1 GEMM: (16x58)->(16x92), 2 K-tiles x 6 N-tiles ----
        {
            v16h A0 = ldA(z1, 64, m16, hh, 0);
            v16h A1 = ldA(z1, 64, m16, hh, 1);
            for (int nt = wid; nt < 6; nt += 4) {   // scalar loop
                v8f acc = {0, 0, 0, 0, 0, 0, 0, 0};
                acc = wmma32(A0, ldB(sW1, 64, m16, hh, 0, nt), acc);
                acc = wmma32(A1, ldB(sW1, 64, m16, hh, 1, nt), acc);
                stC(gbuf, sB1, m16, hh, nt, acc);
            }
        }
        __syncthreads();
        for (int i = tid; i < 16 * 23; i += 128) {
            int m = i / 23, j = i % 23;
            const float* g = &gbuf[m * 256];
            float f1 = ftanh(g[j]), f2 = ftanh(g[23 + j]);
            float tt = sigm(g[46 + j] + g[69 + j]);
            float o  = f1 * (1.0f - tt) + tt * f2;
            _Float16 of = (_Float16)o;
            z2[m * 32 + j]      = of;   // input to CfC2
            zl[m * 96 + 48 + j] = of;   // next-step LSTM h (cols 35..57)
        }
        __syncthreads();

        // ---- CfC2 GEMM: (16x29)->(16x24), 1 K-tile x 2 N-tiles ----
        if (wid < 2) {                              // scalar branch
            v16h A0 = ldA(z2, 32, m16, hh, 0);
            v8f acc = {0, 0, 0, 0, 0, 0, 0, 0};
            acc = wmma32(A0, ldB(sW2, 32, m16, hh, 0, wid), acc);
            stC(gbuf, sB2, m16, hh, wid, acc);
        }
        __syncthreads();
        for (int i = tid; i < 16 * 6; i += 128) {
            int m = i / 6, j = i % 6;
            const float* g = &gbuf[m * 256];
            float f1 = ftanh(g[j]), f2 = ftanh(g[6 + j]);
            float tt = sigm(g[12 + j] + g[18 + j]);
            float o  = f1 * (1.0f - tt) + tt * f2;
            zl[m * 96 + 71 + j] = (_Float16)o;                      // next-step LSTM h (58..63)
            cfcb[(size_t)m * (TSTEPS * 6) + (size_t)t * 6 + j] = o; // motor output (fp32)
        }
        // next-iteration top-of-loop barrier separates these writes from LSTM reads
    }
}

// ---------------------------------------------------------------------------
// Heads: pred = [tanh(y@gw1^T+gb1)@gw2^T+gb2 | same for acc],
//        unc  = softplus(relu(y@uw1^T+ub1)@uw2^T+ub2).  Bandwidth-trivial.
// ---------------------------------------------------------------------------
__global__ __launch_bounds__(256) void heads_kernel(HeadParams P) {
    __shared__ float s[860];
    // offsets: gw1 0, gb1 192, gw2 224, gb2 320 | aw1 323, ab1 515, aw2 547, ab2 643
    //          uw1 646, ub1 742, uw2 758, ub2 854
    for (int i = threadIdx.x; i < 192; i += 256) { s[i] = P.gw1[i]; s[323 + i] = P.aw1[i]; }
    for (int i = threadIdx.x; i < 32;  i += 256) { s[192 + i] = P.gb1[i]; s[515 + i] = P.ab1[i]; }
    for (int i = threadIdx.x; i < 96;  i += 256) {
        s[224 + i] = P.gw2[i]; s[547 + i] = P.aw2[i];
        s[646 + i] = P.uw1[i]; s[758 + i] = P.uw2[i];
    }
    for (int i = threadIdx.x; i < 3;  i += 256) { s[320 + i] = P.gb2[i]; s[643 + i] = P.ab2[i]; }
    for (int i = threadIdx.x; i < 16; i += 256) s[742 + i] = P.ub1[i];
    for (int i = threadIdx.x; i < 6;  i += 256) s[854 + i] = P.ub2[i];
    __syncthreads();

    const size_t TOT = (size_t)BATCH * TSTEPS;
    size_t idx = (size_t)blockIdx.x * 256 + threadIdx.x;
    if (idx >= TOT) return;

    float y[6];
#pragma unroll
    for (int k = 0; k < 6; ++k) y[k] = P.cfc[idx * 6 + k];

    float hbuf[32];
    // gyro
#pragma unroll
    for (int u = 0; u < 32; ++u) {
        float a = s[192 + u];
#pragma unroll
        for (int k = 0; k < 6; ++k) a += s[u * 6 + k] * y[k];
        hbuf[u] = ftanh(a);
    }
#pragma unroll
    for (int o = 0; o < 3; ++o) {
        float a = s[320 + o];
#pragma unroll
        for (int u = 0; u < 32; ++u) a += s[224 + o * 32 + u] * hbuf[u];
        P.out[idx * 6 + o] = a;
    }
    // acc
#pragma unroll
    for (int u = 0; u < 32; ++u) {
        float a = s[515 + u];
#pragma unroll
        for (int k = 0; k < 6; ++k) a += s[323 + u * 6 + k] * y[k];
        hbuf[u] = ftanh(a);
    }
#pragma unroll
    for (int o = 0; o < 3; ++o) {
        float a = s[643 + o];
#pragma unroll
        for (int u = 0; u < 32; ++u) a += s[547 + o * 32 + u] * hbuf[u];
        P.out[idx * 6 + 3 + o] = a;
    }
    // unc
#pragma unroll
    for (int u = 0; u < 16; ++u) {
        float a = s[742 + u];
#pragma unroll
        for (int k = 0; k < 6; ++k) a += s[646 + u * 6 + k] * y[k];
        hbuf[u] = fmaxf(a, 0.0f);
    }
#pragma unroll
    for (int o = 0; o < 6; ++o) {
        float a = s[854 + o];
#pragma unroll
        for (int u = 0; u < 16; ++u) a += s[758 + o * 16 + u] * hbuf[u];
        P.out[TOT * 6 + idx * 6 + o] = (a > 20.0f) ? a : log1pf(__expf(a));
    }
}

extern "C" void kernel_launch(void* const* d_in, const int* in_sizes, int n_in,
                              void* d_out, int out_size, void* d_ws, size_t ws_size,
                              hipStream_t stream) {
    (void)in_sizes; (void)n_in; (void)out_size; (void)ws_size;

    RecParams rp;
    rp.x  = (const float*)d_in[0];
    rp.wi = (const float*)d_in[1];
    rp.wh = (const float*)d_in[2];
    rp.lb = (const float*)d_in[3];
    for (int g = 0; g < 4; ++g) {
        rp.c0w[g] = (const float*)d_in[4 + g];   rp.c0b[g] = (const float*)d_in[8 + g];
        rp.c1w[g] = (const float*)d_in[12 + g];  rp.c1b[g] = (const float*)d_in[16 + g];
        rp.c2w[g] = (const float*)d_in[20 + g];  rp.c2b[g] = (const float*)d_in[24 + g];
    }
    rp.cfc = (float*)d_ws;   // B*T*6 fp32 = 12.6 MB

    HeadParams hp;
    hp.gw1 = (const float*)d_in[28]; hp.gb1 = (const float*)d_in[29];
    hp.gw2 = (const float*)d_in[30]; hp.gb2 = (const float*)d_in[31];
    hp.aw1 = (const float*)d_in[32]; hp.ab1 = (const float*)d_in[33];
    hp.aw2 = (const float*)d_in[34]; hp.ab2 = (const float*)d_in[35];
    hp.uw1 = (const float*)d_in[36]; hp.ub1 = (const float*)d_in[37];
    hp.uw2 = (const float*)d_in[38]; hp.ub2 = (const float*)d_in[39];
    hp.cfc = (const float*)d_ws;
    hp.out = (float*)d_out;

    // 8 batch tiles of 16 rows, sequential T-scan inside the kernel
    rec_kernel<<<dim3(8), dim3(128), 0, stream>>>(rp);
    // B*T = 524288 rows -> 2048 blocks of 256
    heads_kernel<<<dim3((BATCH * TSTEPS) / 256), dim3(256), 0, stream>>>(hp);
}
